// SupConPerAA_26688926777972
// MI455X (gfx1250) — matrix operand order repaired and checked by reading, not compiled, not run on hardware
//
#include <hip/hip_runtime.h>
#include <math.h>

typedef __attribute__((ext_vector_type(16))) _Float16 v16h;
typedef __attribute__((ext_vector_type(8)))  _Float16 v8h;
typedef __attribute__((ext_vector_type(8)))  float    v8f;

#define LCAP  2048
#define NCAP  4096
#define HSIZE 4096
#define MAXR  2560
#define DIM   256
#define INV_T 14.285714285714286f   // 1/0.07

// ---------------------------------------------------------------------------
// Per-sample structure prep: pair list, singleton negatives, row descriptors.
// meta: [0]=n1 [1]=n2 [2]=P [3]=N [4]=M [5]=2M [6]=padRows
// ---------------------------------------------------------------------------
__global__ void supcon_prep(const unsigned char* __restrict__ masks,
                            const int* __restrict__ labelvecs,
                            int i, int B, int L,
                            int* __restrict__ meta,
                            int* __restrict__ idx1, int* __restrict__ idx2,
                            int* __restrict__ rowOff,
                            int* __restrict__ p1, int* __restrict__ p2,
                            int* __restrict__ neg,
                            int* __restrict__ rowdesc, int* __restrict__ rowlab)
{
    __shared__ int sN1, sN2, sP, sN, sM, sTwoM, sPad;
    __shared__ int slv2[LCAP];
    __shared__ int scnt[HSIZE];
    __shared__ int sfirst[HSIZE];
    const int tid = threadIdx.x;
    const int nt  = blockDim.x;

    if (tid == 0) {
        int n1 = 0, n2 = 0;
        for (int l = 0; l < L && l < LCAP; ++l) {
            if (!masks[(size_t)i * L + l])       idx1[n1++] = l;
            if (!masks[(size_t)(B + i) * L + l]) idx2[n2++] = l;
        }
        sN1 = n1; sN2 = n2;
    }
    __syncthreads();
    const int n1 = sN1, n2 = sN2;

    for (int b = tid; b < n2; b += nt) slv2[b] = labelvecs[(size_t)(B + i) * L + b];
    __syncthreads();

    // count label matches per row (row-major pair order, like torch.where)
    for (int a = tid; a < n1; a += nt) {
        const int la = labelvecs[(size_t)i * L + a];
        int c = 0;
        for (int b = 0; b < n2; ++b) c += (la == slv2[b]) ? 1 : 0;
        rowOff[a] = c;
    }
    __syncthreads();
    if (tid == 0) {
        int run = 0;
        for (int a = 0; a < n1; ++a) { int c = rowOff[a]; rowOff[a] = run; run += c; }
        sP = (run > LCAP) ? LCAP : run;
    }
    __syncthreads();
    const int P = sP;

    for (int a = tid; a < n1; a += nt) {
        const int la = labelvecs[(size_t)i * L + a];
        int off = rowOff[a];
        for (int b = 0; b < n2; ++b) {
            if (la == slv2[b]) { if (off < LCAP) { p1[off] = a; p2[off] = b; } ++off; }
        }
    }

    // histogram over cat(lv1, lv2) -> singleton labels in ascending label order
    for (int v = tid; v < HSIZE; v += nt) { scnt[v] = 0; sfirst[v] = 0x7FFFFFFF; }
    __syncthreads();
    const int tot = n1 + n2;
    for (int j = tid; j < tot; j += nt) {
        const int val = (j < n1) ? labelvecs[(size_t)i * L + j] : slv2[j - n1];
        if (val >= 0 && val < HSIZE) { atomicAdd(&scnt[val], 1); atomicMin(&sfirst[val], j); }
    }
    __syncthreads();
    if (tid == 0) {
        int N = 0;
        for (int v = 0; v < HSIZE; ++v)
            if (scnt[v] == 1 && N < NCAP) neg[N++] = sfirst[v];
        int M = P + N;
        int twoM = 2 * M;
        if (P == 0) { M = 0; twoM = 0; N = 0; }
        if (twoM > MAXR) twoM = MAXR & ~15;
        const int pad = (twoM + 15) & ~15;
        sN = N; sM = M; sTwoM = twoM; sPad = pad;
        meta[0] = n1; meta[1] = n2; meta[2] = P; meta[3] = N;
        meta[4] = M;  meta[5] = twoM; meta[6] = pad;
    }
    __syncthreads();
    const int N = sN, M = sM, twoM = sTwoM, pad = sPad;
    if (twoM == 0) return;

    // row descriptors: [0..P) view0 of pairs, [P..P+N) negatives,
    // [M..M+P) view1 of pairs, [M+P..2M) negatives again.
    // labels: pair k -> k ; negative j -> P-1+j  (reference's overlap 'bug')
    for (int k = tid; k < P; k += nt) {
        rowdesc[k] = idx1[p1[k]];
        rowlab[k]  = k;
        if (M + k < MAXR) { rowdesc[M + k] = 0x10000 | idx2[p2[k]]; rowlab[M + k] = k; }
    }
    for (int j = tid; j < N; j += nt) {
        const int g   = neg[j];
        const int dsc = (g < n1) ? idx1[g] : (0x10000 | idx2[g - n1]);
        const int lbl = P - 1 + j;
        if (P + j < MAXR)     { rowdesc[P + j] = dsc;     rowlab[P + j] = lbl; }
        if (M + P + j < MAXR) { rowdesc[M + P + j] = dsc; rowlab[M + P + j] = lbl; }
    }
    for (int r = twoM + tid; r < pad; r += nt) rowlab[r] = -1;
}

// ---------------------------------------------------------------------------
// Gather contrast rows to f16 (row-major [padRows x 256]), zero-pad tail.
// ---------------------------------------------------------------------------
__global__ void supcon_gather(const float* __restrict__ emb, int i, int B, int L,
                              const int* __restrict__ meta,
                              const int* __restrict__ rowdesc,
                              _Float16* __restrict__ Cf16)
{
    const int r    = blockIdx.x;
    const int pad  = meta[6];
    const int twoM = meta[5];
    if (r >= pad) return;
    const int d = threadIdx.x;                 // 0..255 feature dim
    _Float16 val = (_Float16)0.0f;
    if (r < twoM) {
        const int dsc = rowdesc[r];
        const int src = dsc >> 16;
        const int col = dsc & 0xFFFF;
        const size_t base = (size_t)(i + src * B) * DIM * L;
        val = (_Float16)emb[base + (size_t)d * L + col];
    }
    Cf16[(size_t)r * DIM + d] = val;
}

// ---------------------------------------------------------------------------
// Fused Gram(WMMA) + online masked log-softmax SupCon statistics.
// One wave per block; 16 rows per block; K = 256 = 8 x 32.
// ---------------------------------------------------------------------------
__global__ __launch_bounds__(32) void supcon_gemm(
    const _Float16* __restrict__ Cf16,
    const int* __restrict__ rowlab,
    const int* __restrict__ meta,
    float* __restrict__ partials)
{
    const int lane = threadIdx.x;
    const int m16  = lane & 15;
    const int hi   = lane >> 4;                 // half-wave id
    const int twoM = __builtin_amdgcn_readfirstlane(meta[5]);
    const int pad  = __builtin_amdgcn_readfirstlane(meta[6]);
    const int rowBase = blockIdx.x << 4;
    if (pad == 0 || rowBase >= pad) return;     // uniform exit, EXEC untouched

    // labels of this block's rows (D-layout: m = v + hi*8)
    int rlab[8];
#pragma unroll
    for (int v = 0; v < 8; ++v) rlab[v] = rowlab[rowBase + v + (hi << 3)];

    // A panel in registers: ISA 16-bit A layout, lane m: K in {0-7,16-23}+hi*8
    v16h A[8];
#pragma unroll
    for (int kb = 0; kb < 8; ++kb) {
        const _Float16* p = Cf16 + (size_t)(rowBase + m16) * DIM + kb * 32 + (hi << 3);
        const v8h lo  = *(const v8h*)(p);       // K slots 0..7
        const v8h hi8 = *(const v8h*)(p + 16);  // K slots 8..15 (K += 16)
        A[kb] = __builtin_shufflevector(lo, hi8, 0,1,2,3,4,5,6,7,8,9,10,11,12,13,14,15);
    }

    float mRun[8], zRun[8], pSum[8], pCnt[8];
#pragma unroll
    for (int v = 0; v < 8; ++v) { mRun[v] = -3.0e38f; zRun[v] = 0.f; pSum[v] = 0.f; pCnt[v] = 0.f; }

    const int nTiles = pad >> 4;
    for (int nb = 0; nb < nTiles; ++nb) {
        const int colB = (nb << 4) + m16;
        v8f acc = {0.f, 0.f, 0.f, 0.f, 0.f, 0.f, 0.f, 0.f};
#pragma unroll
        for (int kb = 0; kb < 8; ++kb) {
            // B layout: lane n holds column n, K in {0-7,16-23}+hi*8 (matches A)
            const _Float16* q = Cf16 + (size_t)colB * DIM + kb * 32 + (hi << 3);
            const v16h b = __builtin_shufflevector(*(const v8h*)(q), *(const v8h*)(q + 16), 0,1,2,3,4,5,6,7,8,9,10,11,12,13,14,15);
            acc = __builtin_amdgcn_wmma_f32_16x16x32_f16(false, A[kb], false, b,
                                                         (short)0, acc, false, false);
        }
        const bool colOk = colB < twoM;
        const int  clab  = colOk ? rowlab[colB] : -2;
#pragma unroll
        for (int v = 0; v < 8; ++v) {
            const int  row  = rowBase + v + (hi << 3);
            const float x  = acc[v] * INV_T;                  // logit
            float xm = colOk ? x : -3.0e38f;
#pragma unroll
            for (int dm = 1; dm < 16; dm <<= 1) xm = fmaxf(xm, __shfl_xor(xm, dm, 32));
            const float newm = fmaxf(mRun[v], xm);            // row max incl. diagonal
            const bool offd = colOk && (colB != row);         // logits_mask
            const bool pos  = offd && (clab == rlab[v]);      // pos_mask
            float e  = offd ? __expf(x - newm) : 0.f;
            float ps = pos ? x : 0.f;
            float pc = pos ? 1.f : 0.f;
#pragma unroll
            for (int dm = 1; dm < 16; dm <<= 1) {
                e  += __shfl_xor(e,  dm, 32);
                ps += __shfl_xor(ps, dm, 32);
                pc += __shfl_xor(pc, dm, 32);
            }
            zRun[v] = zRun[v] * __expf(mRun[v] - newm) + e;   // online rescale
            pSum[v] += ps;
            pCnt[v] += pc;
            mRun[v] = newm;
        }
    }

    // per-row:  sum_pos log_prob = pSum - pCnt*(m + log z);  contrib = S/pCnt
    float accL = 0.f;
#pragma unroll
    for (int v = 0; v < 8; ++v) {
        const int row = rowBase + v + (hi << 3);
        if (row < twoM && pCnt[v] > 0.f) {
            const float S = pSum[v] - pCnt[v] * (mRun[v] + __logf(zRun[v]));
            accL += S / pCnt[v];
        }
    }
    const float other = __shfl_xor(accL, 16, 32);             // combine half-waves
    if (lane == 0) partials[blockIdx.x] = accL + other;
}

// ---------------------------------------------------------------------------
// Deterministic reductions.
// ---------------------------------------------------------------------------
__global__ void supcon_finalize(const float* __restrict__ partials,
                                const int* __restrict__ meta,
                                float* __restrict__ lossArr, int i)
{
    if (threadIdx.x != 0) return;
    const int twoM = meta[5], pad = meta[6];
    float t = 0.f;
    if (twoM > 0) {
        for (int b = 0; b < (pad >> 4); ++b) t += partials[b];
        t = -t / (float)twoM;                 // -(T/baseT)*mean, T/baseT == 1
    }
    lossArr[i] = t;
}

__global__ void supcon_total(const float* __restrict__ lossArr, int B,
                             float* __restrict__ out)
{
    if (threadIdx.x != 0) return;
    float t = 0.f;
    for (int i = 0; i < B; ++i) t += lossArr[i];
    out[0] = t;
}

// ---------------------------------------------------------------------------
extern "C" void kernel_launch(void* const* d_in, const int* in_sizes, int n_in,
                              void* d_out, int out_size, void* d_ws, size_t ws_size,
                              hipStream_t stream)
{
    const float* emb            = (const float*)d_in[0];   // [2B, 256, L] f32
    const unsigned char* masks  = (const unsigned char*)d_in[1]; // [2B,1,L] bool
    const int* labelvecs        = (const int*)d_in[2];     // [2B, L] int32
    const int L = 1024;
    int B = in_sizes[1] / (2 * L);
    if (B < 1) B = 1;
    if (B > 64) B = 64;

    // workspace carve-out (reused sequentially across samples; stream-ordered)
    char* w = (char*)d_ws;
    auto take = [&](size_t bytes) -> char* {
        char* p = w; w += (bytes + 255) & ~(size_t)255; return p;
    };
    int* meta       = (int*)take(64 * 4);
    int* idx1       = (int*)take((size_t)LCAP * 4);
    int* idx2       = (int*)take((size_t)LCAP * 4);
    int* rowOff     = (int*)take((size_t)LCAP * 4);
    int* p1         = (int*)take((size_t)LCAP * 4);
    int* p2         = (int*)take((size_t)LCAP * 4);
    int* neg        = (int*)take((size_t)NCAP * 4);
    int* rowdesc    = (int*)take((size_t)MAXR * 4);
    int* rowlab     = (int*)take((size_t)MAXR * 4);
    _Float16* Cf16  = (_Float16*)take((size_t)MAXR * DIM * sizeof(_Float16));
    float* partials = (float*)take((MAXR / 16) * 4);
    float* lossArr  = (float*)take(64 * 4);

    for (int i = 0; i < B; ++i) {
        supcon_prep<<<dim3(1), dim3(256), 0, stream>>>(
            masks, labelvecs, i, B, L, meta, idx1, idx2, rowOff, p1, p2, neg,
            rowdesc, rowlab);
        supcon_gather<<<dim3(MAXR), dim3(DIM), 0, stream>>>(
            emb, i, B, L, meta, rowdesc, Cf16);
        supcon_gemm<<<dim3(MAXR / 16), dim3(32), 0, stream>>>(
            Cf16, rowlab, meta, partials);
        supcon_finalize<<<dim3(1), dim3(1), 0, stream>>>(partials, meta, lossArr, i);
    }
    supcon_total<<<dim3(1), dim3(1), 0, stream>>>(lossArr, B, (float*)d_out);
}
